// DebiasedGRUCell_84052509982761
// MI455X (gfx1250) — compile-verified, matrix operand order checked by reading, not run on hardware
//
#include <hip/hip_runtime.h>
#include <hip/hip_bf16.h>

// ---------------------------------------------------------------------------
// DebiasedGRUCell fused kernel for gfx1250 (MI455X).
// B=65536, I=256, H=256.  All GEMMs run on v_wmma_f32_16x16x32_bf16.
// Wave tiling: 32 rows x 64 cols per wave so each weight (B) fragment feeds
// two WMMAs -> 1 global b128-pair per 2 WMMAs in the inner loop.
// ---------------------------------------------------------------------------

#define B_TOTAL 65536
#define I_DIM   256
#define H_DIM   256
#define KA      768      // 2I + H   (r/u GEMM reduction dim)
#define KH      512      // I + H    (h_hat GEMM reduction dim)
#define BM      64       // rows per block
#define AST     776      // LDS row stride (bf16 elems) for activation tile (768 + pad)
#define RST     264      // LDS row stride for r*h_prev tile (256 + pad)

typedef __bf16 bf16_t;
typedef bf16_t v8bf  __attribute__((ext_vector_type(8)));
typedef bf16_t v16bf __attribute__((ext_vector_type(16)));
typedef float  v8f   __attribute__((ext_vector_type(8)));

#define WR_ELEMS (H_DIM * KA)   // 196608
#define WU_ELEMS (H_DIM * KA)   // 196608
#define WH_ELEMS (H_DIM * KH)   // 131072
#define WS_ELEMS (WR_ELEMS + WU_ELEMS + WH_ELEMS)  // 524288 bf16 = 1 MB

// Load one 16-element bf16 WMMA operand fragment.
// CDNA5 16-bit operand layout: lanes 0-15 hold K = k0+0..7 and k0+16..23 of
// row/col (lane&15); lanes 16-31 hold K = k0+8..15 and k0+24..31.
// Caller passes p = base + idx*stride + k0 + (lane>>4)*8.
__device__ __forceinline__ v16bf load_frag(const bf16_t* p) {
  v8bf lo = *(const v8bf*)p;
  v8bf hh = *(const v8bf*)(p + 16);
  v16bf r;
#pragma unroll
  for (int i = 0; i < 8; ++i) { r[i] = lo[i]; r[i + 8] = hh[i]; }
  return r;
}

__device__ __forceinline__ float sigmoid_fast(float x) {
  return 1.0f / (1.0f + __expf(-x));
}
__device__ __forceinline__ float tanh_fast(float x) {
  return 2.0f / (1.0f + __expf(-2.0f * x)) - 1.0f;
}

// ---------------------------------------------------------------------------
// Pre-pass: convert f32 weights to bf16 in workspace (row-major, so each W
// row is directly a K-contiguous WMMA B-operand column).
// ---------------------------------------------------------------------------
__global__ void __launch_bounds__(256)
convert_weights(const float* __restrict__ Wr, const float* __restrict__ Wu,
                const float* __restrict__ Wh, bf16_t* __restrict__ ws) {
  int idx = blockIdx.x * 256 + threadIdx.x;
  if (idx < WR_ELEMS) {
    ws[idx] = (bf16_t)Wr[idx];
  } else if (idx < WR_ELEMS + WU_ELEMS) {
    ws[idx] = (bf16_t)Wu[idx - WR_ELEMS];
  } else if (idx < WS_ELEMS) {
    ws[idx] = (bf16_t)Wh[idx - WR_ELEMS - WU_ELEMS];
  }
}

// ---------------------------------------------------------------------------
// Fused GRU cell. One block = 64 rows x 256 cols of output.
// 8 waves: wave w -> row group (w&1)*32 (2 row tiles), col group (w>>1)*64
// (4 col tiles).  Accumulator tile index = rr*4 + ct.
// ---------------------------------------------------------------------------
__global__ void __launch_bounds__(256)
gru_fused(const float* __restrict__ inputs,   // B x 512
          const float* __restrict__ h_prev,   // B x 256
          const float* __restrict__ score,    // B
          const float* __restrict__ b_r,      // 256
          const float* __restrict__ b_u,      // 256
          const float* __restrict__ b_h,      // 256
          const bf16_t* __restrict__ Wr,      // 256 x 768 bf16
          const bf16_t* __restrict__ Wu,      // 256 x 768 bf16
          const bf16_t* __restrict__ Wh,      // 256 x 512 bf16
          float* __restrict__ out)            // B x 256
{
  extern __shared__ char smem_raw[];
  bf16_t* Alds = (bf16_t*)smem_raw;                                  // BM x AST
  bf16_t* RH   = (bf16_t*)(smem_raw + BM * AST * 2);                 // BM x RST
  float*  sc   = (float*)(smem_raw + BM * AST * 2 + BM * RST * 2);   // BM

  const int tid  = threadIdx.x;
  const int row0 = blockIdx.x * BM;

  // ---- Stage activation tile A = [inputs | h_prev] as bf16 into LDS ----
  for (int i = tid; i < BM * (KA / 4); i += 256) {
    int r = i / (KA / 4);
    int c = (i % (KA / 4)) * 4;
    float4 v;
    if (c < 2 * I_DIM)
      v = *(const float4*)(inputs + (size_t)(row0 + r) * (2 * I_DIM) + c);
    else
      v = *(const float4*)(h_prev + (size_t)(row0 + r) * H_DIM + (c - 2 * I_DIM));
    bf16_t* dst = Alds + r * AST + c;
    dst[0] = (bf16_t)v.x; dst[1] = (bf16_t)v.y;
    dst[2] = (bf16_t)v.z; dst[3] = (bf16_t)v.w;
  }
  if (tid < BM) sc[tid] = score[row0 + tid];
  __syncthreads();

  const int lane  = tid & 31;
  const int wv    = tid >> 5;
  const int rg    = wv & 1;            // row group: rows rg*32 .. rg*32+31
  const int cg    = wv >> 1;           // col group: cols cg*64 .. cg*64+63
  const int nlo   = lane & 15;
  const int hi    = lane >> 4;
  const int mrow0 = rg * 32 + nlo;     // A-operand row, row tile 0
  const int kfo   = hi * 8;            // per-lane K sub-offset

  const v8f vzero = {0.f, 0.f, 0.f, 0.f, 0.f, 0.f, 0.f, 0.f};

  // ---- Phase 1: r and u accumulators over K = 768 ----
  v8f accr[8], accu[8];
#pragma unroll
  for (int t = 0; t < 8; ++t) { accr[t] = vzero; accu[t] = vzero; }

  for (int k0 = 0; k0 < KA; k0 += 32) {
    v16bf a0 = load_frag(Alds + mrow0 * AST + k0 + kfo);
    v16bf a1 = load_frag(Alds + (mrow0 + 16) * AST + k0 + kfo);
#pragma unroll
    for (int ct = 0; ct < 4; ++ct) {
      int n = cg * 64 + ct * 16 + nlo;
      v16bf br = load_frag(Wr + (size_t)n * KA + k0 + kfo);
      v16bf bu = load_frag(Wu + (size_t)n * KA + k0 + kfo);
      accr[ct]     = __builtin_amdgcn_wmma_f32_16x16x32_bf16(
          false, a0, false, br, (short)0, accr[ct],     false, false);
      accr[4 + ct] = __builtin_amdgcn_wmma_f32_16x16x32_bf16(
          false, a1, false, br, (short)0, accr[4 + ct], false, false);
      accu[ct]     = __builtin_amdgcn_wmma_f32_16x16x32_bf16(
          false, a0, false, bu, (short)0, accu[ct],     false, false);
      accu[4 + ct] = __builtin_amdgcn_wmma_f32_16x16x32_bf16(
          false, a1, false, bu, (short)0, accu[4 + ct], false, false);
    }
  }

  // ---- r = sigmoid(accr + b_r); RH[m][n] = bf16(r * h_prev[m][n]) ----
  // C-layout: VGPR v of lane: row = base + (lane>>4)*8 + v, col = nlo + tile.
#pragma unroll
  for (int rr = 0; rr < 2; ++rr) {
#pragma unroll
    for (int ct = 0; ct < 4; ++ct) {
      int n = cg * 64 + ct * 16 + nlo;
      float brn = b_r[n];
#pragma unroll
      for (int v = 0; v < 8; ++v) {
        int ml = rg * 32 + rr * 16 + hi * 8 + v;
        float r  = sigmoid_fast(accr[rr * 4 + ct][v] + brn);
        float hp = (float)Alds[ml * AST + 2 * I_DIM + n];
        RH[ml * RST + n] = (bf16_t)(r * hp);
      }
    }
  }
  __syncthreads();

  // ---- Phase 2: h_hat accumulator over K = 512 ----
  // h_in columns 0..255 alias the staged inputs[:, :256] in Alds.
  v8f acch[8];
#pragma unroll
  for (int t = 0; t < 8; ++t) acch[t] = vzero;

  for (int k0 = 0; k0 < KH; k0 += 32) {
    const bf16_t* ap0 = (k0 < I_DIM)
        ? (Alds + mrow0 * AST + k0 + kfo)
        : (RH + mrow0 * RST + (k0 - I_DIM) + kfo);
    const bf16_t* ap1 = (k0 < I_DIM)
        ? (Alds + (mrow0 + 16) * AST + k0 + kfo)
        : (RH + (mrow0 + 16) * RST + (k0 - I_DIM) + kfo);
    v16bf a0 = load_frag(ap0);
    v16bf a1 = load_frag(ap1);
#pragma unroll
    for (int ct = 0; ct < 4; ++ct) {
      int n = cg * 64 + ct * 16 + nlo;
      v16bf bw = load_frag(Wh + (size_t)n * KH + k0 + kfo);
      acch[ct]     = __builtin_amdgcn_wmma_f32_16x16x32_bf16(
          false, a0, false, bw, (short)0, acch[ct],     false, false);
      acch[4 + ct] = __builtin_amdgcn_wmma_f32_16x16x32_bf16(
          false, a1, false, bw, (short)0, acch[4 + ct], false, false);
    }
  }

  // ---- Final epilogue: u = score*sigmoid(..); out = (1-u)*tanh(..) + u*h_prev
#pragma unroll
  for (int rr = 0; rr < 2; ++rr) {
#pragma unroll
    for (int ct = 0; ct < 4; ++ct) {
      int n = cg * 64 + ct * 16 + nlo;
      float bun = b_u[n];
      float bhn = b_h[n];
#pragma unroll
      for (int v = 0; v < 8; ++v) {
        int ml = rg * 32 + rr * 16 + hi * 8 + v;
        int mg = row0 + ml;
        float u  = sigmoid_fast(accu[rr * 4 + ct][v] + bun) * sc[ml];
        float hh = tanh_fast(acch[rr * 4 + ct][v] + bhn);
        float hp = h_prev[(size_t)mg * H_DIM + n];
        out[(size_t)mg * H_DIM + n] = (1.0f - u) * hh + u * hp;
      }
    }
  }
}

// ---------------------------------------------------------------------------
extern "C" void kernel_launch(void* const* d_in, const int* in_sizes, int n_in,
                              void* d_out, int out_size, void* d_ws, size_t ws_size,
                              hipStream_t stream) {
  (void)in_sizes; (void)n_in; (void)out_size; (void)ws_size;

  const float* inputs = (const float*)d_in[0];
  const float* h_prev = (const float*)d_in[1];
  const float* score  = (const float*)d_in[2];
  const float* W_r    = (const float*)d_in[3];
  const float* b_r    = (const float*)d_in[4];
  const float* W_u    = (const float*)d_in[5];
  const float* b_u    = (const float*)d_in[6];
  const float* W_h    = (const float*)d_in[7];
  const float* b_h    = (const float*)d_in[8];
  float* out          = (float*)d_out;

  bf16_t* ws          = (bf16_t*)d_ws;          // 1 MB of bf16 weights
  const bf16_t* Wr_bf = ws;
  const bf16_t* Wu_bf = ws + WR_ELEMS;
  const bf16_t* Wh_bf = ws + WR_ELEMS + WU_ELEMS;

  // Convert weights f32 -> bf16 (L2-resident; ~1 MB).
  convert_weights<<<(WS_ELEMS + 255) / 256, 256, 0, stream>>>(W_r, W_u, W_h, ws);

  const int smem_bytes = BM * AST * 2 + BM * RST * 2 + BM * 4;  // ~133 KB
  // Allow >64KB dynamic LDS (WGP supports up to 320KB).
  hipFuncSetAttribute(reinterpret_cast<const void*>(&gru_fused),
                      hipFuncAttributeMaxDynamicSharedMemorySize, smem_bytes);

  gru_fused<<<B_TOTAL / BM, 256, smem_bytes, stream>>>(
      inputs, h_prev, score, b_r, b_u, b_h, Wr_bf, Wu_bf, Wh_bf, out);
}
